// KQAlignedAttention_85074712199240
// MI455X (gfx1250) — matrix operand order repaired and checked by reading, not compiled, not run on hardware
//
#include <hip/hip_runtime.h>

// MI455X / gfx1250: wave32, WMMA 16x16x32 bf16 (fp32 accum),
// async global->LDS tile staging (ASYNCcnt), DPP row reductions.
typedef __bf16 bf16_t;
typedef __attribute__((ext_vector_type(16))) __bf16 v16bf;
typedef __attribute__((ext_vector_type(8)))  __bf16 v8bf;
typedef __attribute__((ext_vector_type(8)))  float  v8f;

#define NUM_HEADS 12
#define HEAD_DIM  64
#define DIM       768
#define BATCH     8
#define SEQ       1024
#define BN        (BATCH * SEQ)

__device__ __forceinline__ v8f v8f_zero() {
  v8f r;
#pragma unroll
  for (int i = 0; i < 8; ++i) r[i] = 0.f;
  return r;
}

__device__ __forceinline__ v8f wmma_bf16(v16bf a, v16bf b, v8f c) {
  // (neg_a, A, neg_b, B, c_mod, C, reuse_a, reuse_b)
  return __builtin_amdgcn_wmma_f32_16x16x32_bf16(false, a, false, b, (short)0, c,
                                                 false, false);
}

// Async DMA: 16 bytes/lane global -> LDS, tracked by ASYNCcnt (no VGPR data).
// LDS address = low 32 bits of the generic shared pointer (ISA: addr[31:0]).
__device__ __forceinline__ void async_copy_b128(void* lds_dst, const void* gsrc) {
  unsigned lds_off = (unsigned)(unsigned long long)lds_dst;
  asm volatile("global_load_async_to_lds_b128 %0, %1, off"
               :: "v"(lds_off), "v"(gsrc)
               : "memory");
}
__device__ __forceinline__ void wait_async_0() {
  asm volatile("s_wait_asynccnt 0x0" ::: "memory");
}
__device__ __forceinline__ void wait_async_3() {
  asm volatile("s_wait_asynccnt 0x3" ::: "memory");
}

// DPP row_ror:N within 16-lane rows (matches the C/D-layout lane halves).
template <int N>
__device__ __forceinline__ float dpp_ror(float x) {
  int r = __builtin_amdgcn_update_dpp(0, __builtin_bit_cast(int, x),
                                      0x120 + N, 0xF, 0xF, true);
  return __builtin_bit_cast(float, r);
}
__device__ __forceinline__ float row_max16(float x) {
  x = fmaxf(x, dpp_ror<1>(x));
  x = fmaxf(x, dpp_ror<2>(x));
  x = fmaxf(x, dpp_ror<4>(x));
  x = fmaxf(x, dpp_ror<8>(x));
  return x;
}
__device__ __forceinline__ float row_sum16(float x) {
  x += dpp_ror<1>(x);
  x += dpp_ror<2>(x);
  x += dpp_ror<4>(x);
  x += dpp_ror<8>(x);
  return x;
}

// A fragment (16x32, M x K), bf16. Lanes 0-15 -> K base 0, lanes 16-31 -> K
// base 8; VGPR0-3 = K kb..kb+7, VGPR4-7 = K kb+16..kb+23.
__device__ __forceinline__ v16bf frag_a(const bf16_t* base, int stride) {
  const int lane = threadIdx.x & 31;
  const int m = lane & 15;
  const int kb = (lane >> 4) << 3;
  const bf16_t* p = base + m * stride;
  union { v16bf v; v8bf h[2]; } u;
  u.h[0] = *(const v8bf*)(p + kb);
  u.h[1] = *(const v8bf*)(p + kb + 16);
  return u.v;
}

// B fragment (32x16, K x N) from an N-major LDS tile Bt[n][k]:
// lanes 0-15 -> K base 0, lanes 16-31 -> K base 8; K kb..kb+7, kb+16..kb+23.
__device__ __forceinline__ v16bf frag_b(const bf16_t* baseT, int stride) {
  const int lane = threadIdx.x & 31;
  const int nn = lane & 15;
  const int kb = (lane >> 4) << 3;
  const bf16_t* p = baseT + nn * stride + kb;
  union { v16bf v; v8bf h[2]; } u;
  u.h[0] = *(const v8bf*)(p);
  u.h[1] = *(const v8bf*)(p + 16);
  return u.v;
}

// ---------------- prep kernels ----------------
__global__ void __launch_bounds__(256) cvt_kernel(const float* __restrict__ in,
                                                  bf16_t* __restrict__ out, int count) {
  int i = blockIdx.x * blockDim.x + threadIdx.x;
  int stride = gridDim.x * blockDim.x;
  for (; i < count; i += stride) out[i] = (bf16_t)in[i];
}

__global__ void __launch_bounds__(256) prep_w_kernel(const float* __restrict__ W,
                                                     bf16_t* __restrict__ Wb,
                                                     bf16_t* __restrict__ WTb) {
  int c = blockIdx.x * blockDim.x + threadIdx.x;
  int r = blockIdx.y;
  if (c < DIM) {
    float v = W[r * DIM + c];
    Wb[r * DIM + c] = (bf16_t)v;
    WTb[c * DIM + r] = (bf16_t)v;
  }
}

// ---------------- GEMM: C = [-(if NEGATE)] sum_p A_p @ B_p^T ----------------
// A: M x K row-major bf16.  B: N x K row-major bf16 (operand is B^T).
// 256 thr (8 waves), tile 64(M) x 128(N), K-step 32, double-buffered LDS with
// async global->LDS staging pipelined against WMMA.
template <int NPAIR, bool OUT_BF16, bool NEGATE>
__global__ void __launch_bounds__(256)
gemm_kernel(const bf16_t* __restrict__ A0, const bf16_t* __restrict__ B0,
            const bf16_t* __restrict__ A1, const bf16_t* __restrict__ B1,
            void* __restrict__ Cout, int M, int N, int K) {
  __shared__ bf16_t Als[2][64 * 40];
  __shared__ bf16_t Bls[2][128 * 40];
  const int tid = threadIdx.x;
  const int lane = tid & 31, wave = tid >> 5;
  const int wm = wave >> 1, wn = wave & 1;
  const int m0 = blockIdx.y * 64;
  const int n0 = blockIdx.x * 128;
  const int arow = tid >> 2, akc = (tid & 3) << 3;
  const int brow = tid >> 1, bkc = (tid & 1) << 4;

  v8f acc[4];
#pragma unroll
  for (int j = 0; j < 4; ++j) acc[j] = v8f_zero();

  for (int p = 0; p < NPAIR; ++p) {
    const bf16_t* A = (p == 0) ? A0 : A1;
    const bf16_t* B = (p == 0) ? B0 : B1;
    const bf16_t* Arow = A + (size_t)(m0 + arow) * K + akc;
    const bf16_t* Brow = B + (size_t)(n0 + brow) * K + bkc;

    // stage tile 0 into buffer 0
    async_copy_b128(&Als[0][arow * 40 + akc], Arow);
    async_copy_b128(&Bls[0][brow * 40 + bkc], Brow);
    async_copy_b128(&Bls[0][brow * 40 + bkc + 8], Brow + 8);
    wait_async_0();
    __syncthreads();

    for (int k0 = 0; k0 < K; k0 += 32) {
      const int cur = (k0 >> 5) & 1;
      if (k0 + 32 < K) {  // prefetch next tile into the alternate buffer
        async_copy_b128(&Als[cur ^ 1][arow * 40 + akc], Arow + k0 + 32);
        async_copy_b128(&Bls[cur ^ 1][brow * 40 + bkc], Brow + k0 + 32);
        async_copy_b128(&Bls[cur ^ 1][brow * 40 + bkc + 8], Brow + k0 + 40);
      }
      v16bf af = frag_a(&Als[cur][16 * wm * 40], 40);
#pragma unroll
      for (int j = 0; j < 4; ++j) {
        v16bf bf = frag_b(&Bls[cur][(64 * wn + 16 * j) * 40], 40);
        acc[j] = wmma_bf16(af, bf, acc[j]);
      }
      wait_async_3();  // current tile's copies done; next tile may stay in flight
      __syncthreads();
    }
  }

  const int nn = lane & 15;
  const int rb = (lane >> 4) << 3;
#pragma unroll
  for (int j = 0; j < 4; ++j) {
#pragma unroll
    for (int v = 0; v < 8; ++v) {
      const int row = m0 + 16 * wm + rb + v;
      const int col = n0 + 64 * wn + 16 * j + nn;
      float val = NEGATE ? -acc[j][v] : acc[j][v];
      if constexpr (OUT_BF16)
        ((bf16_t*)Cout)[(size_t)row * N + col] = (bf16_t)val;
      else
        ((float*)Cout)[(size_t)row * N + col] = val;
    }
  }
}

// ---------------- attention pass 1: O1 = softmax(Q K^T * beta) @ K ----------
__global__ void __launch_bounds__(128)
attn_pass1(const bf16_t* __restrict__ Qb, const bf16_t* __restrict__ Kb,
           const float* __restrict__ betas, bf16_t* __restrict__ O1,
           float* __restrict__ mbuf, float* __restrict__ lbuf) {
  __shared__ bf16_t Qs[64 * 72];      // [q][z]
  __shared__ bf16_t Ks[32 * 72];      // [k][z]  (B operand of S)
  __shared__ bf16_t KTs[64 * 40];     // [z][k]  (B operand of O1)
  __shared__ bf16_t Ps[4 * 16 * 40];  // wave-private P scratch [q][k]
  const int tid = threadIdx.x, lane = tid & 31, wave = tid >> 5;
  const int b = blockIdx.z, h = blockIdx.y;
  const int q0 = blockIdx.x * 64;
  const float beta = betas[h];
  const size_t base = (size_t)b * SEQ * DIM + (size_t)h * HEAD_DIM;
  const int nn = lane & 15;
  const int rb = (lane >> 4) << 3;

  {  // load Q tile 64x64
    const int row = tid >> 1, half = tid & 1;
    const bf16_t* src = Qb + base + (size_t)(q0 + row) * DIM + half * 32;
#pragma unroll
    for (int i = 0; i < 4; ++i)
      *(v8bf*)(&Qs[row * 72 + half * 32 + 8 * i]) = *(const v8bf*)(src + 8 * i);
  }
  __syncthreads();
  v16bf qfa[2];
  qfa[0] = frag_a(&Qs[16 * wave * 72], 72);
  qfa[1] = frag_a(&Qs[16 * wave * 72 + 32], 72);

  v8f o1[4];
#pragma unroll
  for (int j = 0; j < 4; ++j) o1[j] = v8f_zero();
  float mrow[8], lrow[8];
#pragma unroll
  for (int v = 0; v < 8; ++v) { mrow[v] = -1e30f; lrow[v] = 0.f; }
  bf16_t* Pw = &Ps[wave * 16 * 40];

  for (int kt = 0; kt < SEQ; kt += 32) {
    __syncthreads();
    {  // load K tile 32x64 natural + transposed
      const int row = tid >> 2, qd = tid & 3;
#pragma unroll
      for (int i = 0; i < 2; ++i) {
        const int z = qd * 8 + 32 * i;
        v8bf val = *(const v8bf*)(Kb + base + (size_t)(kt + row) * DIM + z);
        *(v8bf*)(&Ks[row * 72 + z]) = val;
#pragma unroll
        for (int e = 0; e < 8; ++e) KTs[(z + e) * 40 + row] = val[e];
      }
    }
    __syncthreads();

    v8f s[2];
#pragma unroll
    for (int jt = 0; jt < 2; ++jt) {
      v8f sa = v8f_zero();
#pragma unroll
      for (int zs = 0; zs < 2; ++zs)
        sa = wmma_bf16(qfa[zs], frag_b(&Ks[16 * jt * 72 + zs * 32], 72), sa);
      s[jt] = sa;
    }

    // online softmax over the 32 new columns (DPP row reductions), write P
#pragma unroll
    for (int v = 0; v < 8; ++v) {
      float a = s[0][v] * beta, c = s[1][v] * beta;
      float mx = row_max16(fmaxf(a, c));
      float mnew = fmaxf(mrow[v], mx);
      float sc = __expf(mrow[v] - mnew);
      float p0 = __expf(a - mnew), p1 = __expf(c - mnew);
      float rs = row_sum16(p0 + p1);
      lrow[v] = lrow[v] * sc + rs;
      mrow[v] = mnew;
#pragma unroll
      for (int j = 0; j < 4; ++j) o1[j][v] *= sc;
      Pw[(rb + v) * 40 + nn] = (bf16_t)p0;
      Pw[(rb + v) * 40 + 16 + nn] = (bf16_t)p1;
    }

    v16bf pf = frag_a(Pw, 40);
#pragma unroll
    for (int j = 0; j < 4; ++j)
      o1[j] = wmma_bf16(pf, frag_b(&KTs[16 * j * 40], 40), o1[j]);
  }

  float rl[8];
#pragma unroll
  for (int v = 0; v < 8; ++v) rl[v] = 1.f / lrow[v];
#pragma unroll
  for (int j = 0; j < 4; ++j)
#pragma unroll
    for (int v = 0; v < 8; ++v) {
      const int q = q0 + 16 * wave + rb + v;
      O1[((size_t)(b * SEQ + q)) * DIM + h * HEAD_DIM + 16 * j + nn] =
          (bf16_t)(o1[j][v] * rl[v]);
    }
  if (nn == 0) {
#pragma unroll
    for (int v = 0; v < 8; ++v) {
      const int q = q0 + 16 * wave + rb + v;
      mbuf[((size_t)(b * NUM_HEADS + h)) * SEQ + q] = mrow[v];
      lbuf[((size_t)(b * NUM_HEADS + h)) * SEQ + q] = lrow[v];
    }
  }
}

// ---------------- attention pass 2: O2 = softmax(...)^T @ Q ----------------
__global__ void __launch_bounds__(128)
attn_pass2(const bf16_t* __restrict__ Qb, const bf16_t* __restrict__ Kb,
           const float* __restrict__ betas, const float* __restrict__ mbuf,
           const float* __restrict__ lbuf, bf16_t* __restrict__ O2) {
  __shared__ bf16_t Kts[64 * 72];     // [k][z]  (A operand)
  __shared__ bf16_t Qs[32 * 72];      // [q][z]  (B operand of S^T)
  __shared__ bf16_t QTs[64 * 40];     // [z][q]  (B operand of O2)
  __shared__ bf16_t Ps[4 * 16 * 40];  // wave-private P^T scratch [k][q]
  const int tid = threadIdx.x, lane = tid & 31, wave = tid >> 5;
  const int b = blockIdx.z, h = blockIdx.y;
  const int k0 = blockIdx.x * 64;
  const float beta = betas[h];
  const size_t base = (size_t)b * SEQ * DIM + (size_t)h * HEAD_DIM;
  const float* mrow_g = mbuf + ((size_t)(b * NUM_HEADS + h)) * SEQ;
  const float* lrow_g = lbuf + ((size_t)(b * NUM_HEADS + h)) * SEQ;
  const int nn = lane & 15;
  const int rb = (lane >> 4) << 3;

  {  // load K tile 64x64
    const int row = tid >> 1, half = tid & 1;
    const bf16_t* src = Kb + base + (size_t)(k0 + row) * DIM + half * 32;
#pragma unroll
    for (int i = 0; i < 4; ++i)
      *(v8bf*)(&Kts[row * 72 + half * 32 + 8 * i]) = *(const v8bf*)(src + 8 * i);
  }
  __syncthreads();
  v16bf kfa[2];
  kfa[0] = frag_a(&Kts[16 * wave * 72], 72);
  kfa[1] = frag_a(&Kts[16 * wave * 72 + 32], 72);

  v8f o2[4];
#pragma unroll
  for (int j = 0; j < 4; ++j) o2[j] = v8f_zero();
  bf16_t* Pw = &Ps[wave * 16 * 40];

  for (int qt = 0; qt < SEQ; qt += 32) {
    __syncthreads();
    {  // load Q tile 32x64 natural + transposed
      const int row = tid >> 2, qd = tid & 3;
#pragma unroll
      for (int i = 0; i < 2; ++i) {
        const int z = qd * 8 + 32 * i;
        v8bf val = *(const v8bf*)(Qb + base + (size_t)(qt + row) * DIM + z);
        *(v8bf*)(&Qs[row * 72 + z]) = val;
#pragma unroll
        for (int e = 0; e < 8; ++e) QTs[(z + e) * 40 + row] = val[e];
      }
    }
    __syncthreads();

#pragma unroll
    for (int jt = 0; jt < 2; ++jt) {
      v8f sa = v8f_zero();
#pragma unroll
      for (int zs = 0; zs < 2; ++zs)
        sa = wmma_bf16(kfa[zs], frag_b(&Qs[16 * jt * 72 + zs * 32], 72), sa);
      const int q = qt + 16 * jt + nn;
      const float mq = mrow_g[q];
      const float rlq = 1.f / lrow_g[q];
#pragma unroll
      for (int v = 0; v < 8; ++v) {
        float p = __expf(sa[v] * beta - mq) * rlq;
        Pw[(rb + v) * 40 + 16 * jt + nn] = (bf16_t)p;
      }
    }

    v16bf pf = frag_a(Pw, 40);
#pragma unroll
    for (int j = 0; j < 4; ++j)
      o2[j] = wmma_bf16(pf, frag_b(&QTs[16 * j * 40], 40), o2[j]);
  }

#pragma unroll
  for (int j = 0; j < 4; ++j)
#pragma unroll
    for (int v = 0; v < 8; ++v) {
      const int k = k0 + 16 * wave + rb + v;
      O2[((size_t)(b * SEQ + k)) * DIM + h * HEAD_DIM + 16 * j + nn] =
          (bf16_t)o2[j][v];
    }
}

// ---------------- launch ----------------
extern "C" void kernel_launch(void* const* d_in, const int* in_sizes, int n_in,
                              void* d_out, int out_size, void* d_ws, size_t ws_size,
                              hipStream_t stream) {
  (void)in_sizes; (void)n_in; (void)out_size; (void)ws_size;
  const float* x = (const float*)d_in[0];
  const float* Wq = (const float*)d_in[1];
  const float* Wk = (const float*)d_in[2];
  const float* betas = (const float*)d_in[3];

  char* ws = (char*)d_ws;
  size_t off = 0;
  auto take = [&](size_t bytes) {
    char* p = ws + off;
    off += (bytes + 255) & ~(size_t)255;
    return p;
  };
  bf16_t* xb   = (bf16_t*)take((size_t)BN * DIM * 2);
  bf16_t* Wqb  = (bf16_t*)take((size_t)DIM * DIM * 2);
  bf16_t* Wkb  = (bf16_t*)take((size_t)DIM * DIM * 2);
  bf16_t* WqTb = (bf16_t*)take((size_t)DIM * DIM * 2);
  bf16_t* WkTb = (bf16_t*)take((size_t)DIM * DIM * 2);
  bf16_t* Qb   = (bf16_t*)take((size_t)BN * DIM * 2);
  bf16_t* Kb   = (bf16_t*)take((size_t)BN * DIM * 2);
  bf16_t* O1   = (bf16_t*)take((size_t)BN * DIM * 2);
  bf16_t* O2   = (bf16_t*)take((size_t)BN * DIM * 2);
  float* mbuf  = (float*)take((size_t)BATCH * NUM_HEADS * SEQ * 4);
  float* lbuf  = (float*)take((size_t)BATCH * NUM_HEADS * SEQ * 4);

  hipLaunchKernelGGL(cvt_kernel, dim3(1024), dim3(256), 0, stream, x, xb, BN * DIM);
  hipLaunchKernelGGL(prep_w_kernel, dim3(3, DIM), dim3(256), 0, stream, Wq, Wqb, WqTb);
  hipLaunchKernelGGL(prep_w_kernel, dim3(3, DIM), dim3(256), 0, stream, Wk, Wkb, WkTb);

  dim3 ggrid(DIM / 128, BN / 64);
  hipLaunchKernelGGL((gemm_kernel<1, true, false>), ggrid, dim3(256), 0, stream,
                     xb, Wqb, xb, Wqb, (void*)Qb, BN, DIM, DIM);
  hipLaunchKernelGGL((gemm_kernel<1, true, false>), ggrid, dim3(256), 0, stream,
                     xb, Wkb, xb, Wkb, (void*)Kb, BN, DIM, DIM);

  hipLaunchKernelGGL(attn_pass1, dim3(SEQ / 64, NUM_HEADS, BATCH), dim3(128), 0,
                     stream, Qb, Kb, betas, O1, mbuf, lbuf);
  hipLaunchKernelGGL(attn_pass2, dim3(SEQ / 64, NUM_HEADS, BATCH), dim3(128), 0,
                     stream, Qb, Kb, betas, mbuf, lbuf, O2);

  hipLaunchKernelGGL((gemm_kernel<2, false, true>), ggrid, dim3(256), 0, stream,
                     O1, WqTb, O2, WkTb, d_out, BN, DIM, DIM);
}
